// Model_78520592106021
// MI455X (gfx1250) — compile-verified, matrix operand order checked by reading, not run on hardware
//
#include <hip/hip_runtime.h>

// ---------------------------------------------------------------------------
// CDNA5 (gfx1250, wave32) music-VAE forward pass.
//   * Large GEMMs (input-gate precompute, MLPs): bf16 WMMA, 64x64x64 block
//     tiles, vectorized 16B LDS staging, f32 accumulate.
//   * LSTM recurrence: one WMMA kernel per timestep (32 blocks x 8 waves),
//     h state ping-pongs between bf16 buffers.
//   * Tiny-N projections (1/9/39): scalar float4 dot kernels.
// ---------------------------------------------------------------------------

typedef __attribute__((ext_vector_type(16))) __bf16 bf16x16;
typedef __attribute__((ext_vector_type(8)))  __bf16 bf16x8;
typedef __attribute__((ext_vector_type(8)))  float  f32x8;

#define SHUF16(lo, hi) __builtin_shufflevector(lo, hi, 0,1,2,3,4,5,6,7,8,9,10,11,12,13,14,15)

__device__ __forceinline__ float sigf(float x) { return 1.0f / (1.0f + __expf(-x)); }

// ---------------- elementwise helpers ----------------
__global__ void k_f2bf(const float* __restrict__ in, __bf16* __restrict__ out, int n) {
  int i = blockIdx.x * blockDim.x + threadIdx.x;
  if (i < n) out[i] = (__bf16)in[i];
}
__global__ void k_zero_f(float* p, int n) {
  int i = blockIdx.x * blockDim.x + threadIdx.x;
  if (i < n) p[i] = 0.0f;
}
__global__ void k_zero_bf(__bf16* p, int n) {
  int i = blockIdx.x * blockDim.x + threadIdx.x;
  if (i < n) p[i] = (__bf16)0.0f;
}
__global__ void k_copy_f(const float* __restrict__ a, float* __restrict__ b, int n) {
  int i = blockIdx.x * blockDim.x + threadIdx.x;
  if (i < n) b[i] = a[i];
}
__global__ void k_add(const float* __restrict__ a, const float* __restrict__ b,
                      float* __restrict__ c, int n) {
  int i = blockIdx.x * blockDim.x + threadIdx.x;
  if (i < n) c[i] = a[i] + b[i];
}

// ---------------- generic WMMA GEMM ----------------
// C[M,N] = act( A[M,K](f32, strided; row r = t*64 + b) * W[N,K](bf16)^T + bias )
// Requires: M%64==0, N%64==0, K%64==0.  grid = (N/64, M/64), 256 threads.
__global__ void k_gemm(const float* __restrict__ A, long sT, long sB,
                       const __bf16* __restrict__ W, const float* __restrict__ bias,
                       float* __restrict__ C, int K, int relu) {
  __shared__ __align__(16) __bf16 As[64][72];   // 64 rows x 64 k (+8 pad), 144B row stride
  __shared__ __align__(16) __bf16 Bs[64][72];
  const int tid  = threadIdx.x;
  const int lane = tid & 31;
  const int w    = tid >> 5;
  const int ct   = w & 3;          // column tile 0..3
  const int rt0  = (w >> 2) * 2;   // row tiles rt0, rt0+1
  const int N    = gridDim.x * 64;
  const long rowBase = (long)blockIdx.y * 64;
  const int  colBase = blockIdx.x * 64;
  const int  m  = lane & 15;
  const int  kb = (lane >> 4) * 8;

  f32x8 acc0 = {}, acc1 = {};
  for (int k0 = 0; k0 < K; k0 += 64) {
    __syncthreads();
#pragma unroll
    for (int cc = 0; cc < 2; ++cc) {              // 2 chunks of 8 elems per matrix per thread
      int ci = tid + cc * 256;                    // chunk id 0..511
      int r = ci >> 3, c8 = (ci & 7) * 8;
      long rg = rowBase + r;
      long t = rg >> 6, b = rg & 63;
      const float* ap = A + b * sB + t * sT + (long)(k0 + c8);
      float4 a0 = *(const float4*)ap;
      float4 a1 = *(const float4*)(ap + 4);
      bf16x8 v;
      v[0] = (__bf16)a0.x; v[1] = (__bf16)a0.y; v[2] = (__bf16)a0.z; v[3] = (__bf16)a0.w;
      v[4] = (__bf16)a1.x; v[5] = (__bf16)a1.y; v[6] = (__bf16)a1.z; v[7] = (__bf16)a1.w;
      *(bf16x8*)&As[r][c8] = v;
      *(bf16x8*)&Bs[r][c8] = *(const bf16x8*)&W[(long)(colBase + r) * K + (k0 + c8)];
    }
    __syncthreads();
#pragma unroll
    for (int kc = 0; kc < 64; kc += 32) {
      bf16x8 a0l = *(const bf16x8*)&As[rt0 * 16 + m][kc + kb];
      bf16x8 a0h = *(const bf16x8*)&As[rt0 * 16 + m][kc + kb + 16];
      bf16x8 a1l = *(const bf16x8*)&As[rt0 * 16 + 16 + m][kc + kb];
      bf16x8 a1h = *(const bf16x8*)&As[rt0 * 16 + 16 + m][kc + kb + 16];
      bf16x8 bl  = *(const bf16x8*)&Bs[ct * 16 + m][kc + kb];
      bf16x8 bh  = *(const bf16x8*)&Bs[ct * 16 + m][kc + kb + 16];
      bf16x16 af0 = SHUF16(a0l, a0h);
      bf16x16 af1 = SHUF16(a1l, a1h);
      bf16x16 bfr = SHUF16(bl,  bh);
      acc0 = __builtin_amdgcn_wmma_f32_16x16x32_bf16(false, af0, false, bfr, (short)0, acc0, false, false);
      acc1 = __builtin_amdgcn_wmma_f32_16x16x32_bf16(false, af1, false, bfr, (short)0, acc1, false, false);
    }
  }

  const int n = colBase + ct * 16 + (lane & 15);
  const float bv = bias ? bias[n] : 0.0f;
  const int mrow = (lane >> 4) * 8;
#pragma unroll
  for (int v = 0; v < 8; ++v) {
    long m0 = rowBase + rt0 * 16 + v + mrow;
    float x0 = acc0[v] + bv;
    float x1 = acc1[v] + bv;
    if (relu) { x0 = fmaxf(x0, 0.0f); x1 = fmaxf(x1, 0.0f); }
    C[m0 * N + n] = x0;
    C[(m0 + 16) * N + n] = x1;
  }
}

// ---------------- one LSTM timestep ----------------
// gates = gin_t + h_in @ Whh^T + bhh ; cell update ; write h/c.
// grid = 32 blocks (16 columns of H each), 256 threads (8 waves).
__global__ void k_lstm_step(const __bf16* __restrict__ h_in, __bf16* __restrict__ h_out,
                            float* __restrict__ c_buf, const __bf16* __restrict__ Whh,
                            const float* __restrict__ gin_t, const float* __restrict__ bhh,
                            float* __restrict__ h_store) {
  __shared__ __align__(16) __bf16 Ah[64][72];
  __shared__ __align__(16) __bf16 Bw[64][72];
  __shared__ float  Gg[4][64][16];
  const int tid  = threadIdx.x;
  const int lane = tid & 31;
  const int w    = tid >> 5;
  const int gidx = w & 3;          // which gate (i,f,g,o)
  const int rt0  = (w >> 2) * 2;   // batch-row tiles rt0, rt0+1
  const int n0   = blockIdx.x * 16;
  const int m  = lane & 15;
  const int kb = (lane >> 4) * 8;

  f32x8 acc0 = {}, acc1 = {};
  for (int k0 = 0; k0 < 512; k0 += 64) {
    __syncthreads();
#pragma unroll
    for (int cc = 0; cc < 2; ++cc) {
      int ci = tid + cc * 256;
      int r = ci >> 3, c8 = (ci & 7) * 8;
      *(bf16x8*)&Ah[r][c8] = *(const bf16x8*)&h_in[r * 512 + k0 + c8];
      int wr = (r >> 4) * 512 + n0 + (r & 15);
      *(bf16x8*)&Bw[r][c8] = *(const bf16x8*)&Whh[(long)wr * 512 + k0 + c8];
    }
    __syncthreads();
#pragma unroll
    for (int kc = 0; kc < 64; kc += 32) {
      bf16x8 a0l = *(const bf16x8*)&Ah[rt0 * 16 + m][kc + kb];
      bf16x8 a0h = *(const bf16x8*)&Ah[rt0 * 16 + m][kc + kb + 16];
      bf16x8 a1l = *(const bf16x8*)&Ah[rt0 * 16 + 16 + m][kc + kb];
      bf16x8 a1h = *(const bf16x8*)&Ah[rt0 * 16 + 16 + m][kc + kb + 16];
      bf16x8 bl  = *(const bf16x8*)&Bw[gidx * 16 + m][kc + kb];
      bf16x8 bh  = *(const bf16x8*)&Bw[gidx * 16 + m][kc + kb + 16];
      bf16x16 af0 = SHUF16(a0l, a0h);
      bf16x16 af1 = SHUF16(a1l, a1h);
      bf16x16 bfr = SHUF16(bl,  bh);
      acc0 = __builtin_amdgcn_wmma_f32_16x16x32_bf16(false, af0, false, bfr, (short)0, acc0, false, false);
      acc1 = __builtin_amdgcn_wmma_f32_16x16x32_bf16(false, af1, false, bfr, (short)0, acc1, false, false);
    }
  }

  const int mrow = (lane >> 4) * 8;
  __syncthreads();
#pragma unroll
  for (int v = 0; v < 8; ++v) {
    Gg[gidx][rt0 * 16 + v + mrow][m]      = acc0[v];
    Gg[gidx][rt0 * 16 + 16 + v + mrow][m] = acc1[v];
  }
  __syncthreads();

#pragma unroll
  for (int q = 0; q < 4; ++q) {
    int e = tid + q * 256;          // 64 rows x 16 cols = 1024 cells
    int b = e >> 4, col = e & 15;
    int cg = n0 + col;
    float ii = Gg[0][b][col] + gin_t[b * 2048 + cg]        + bhh[cg];
    float ff = Gg[1][b][col] + gin_t[b * 2048 + 512 + cg]  + bhh[512 + cg];
    float gg = Gg[2][b][col] + gin_t[b * 2048 + 1024 + cg] + bhh[1024 + cg];
    float oo = Gg[3][b][col] + gin_t[b * 2048 + 1536 + cg] + bhh[1536 + cg];
    float cold = c_buf[b * 512 + cg];
    float cn = sigf(ff) * cold + sigf(ii) * tanhf(gg);
    float hv = sigf(oo) * tanhf(cn);
    c_buf[b * 512 + cg]  = cn;
    h_out[b * 512 + cg]  = (__bf16)hv;
    h_store[b * 512 + cg] = hv;
  }
}

// ---------------- small-N projection: out = A[M,K] @ W[N,K]^T + bias ----------------
// out index = b*sb + t*st + j*sn, with row r = t*64 + b.
__global__ void k_proj(const float* __restrict__ A, const float* __restrict__ W,
                       const float* __restrict__ bias, float* __restrict__ out,
                       int M, int N, int K, int sb, int st, int sn) {
  int idx = blockIdx.x * blockDim.x + threadIdx.x;
  if (idx >= M * N) return;
  int r = idx / N, j = idx % N;
  int t = r >> 6, b = r & 63;
  const float4* a4 = (const float4*)(A + (long)r * K);
  const float4* w4 = (const float4*)(W + (long)j * K);
  float s = 0.0f;
  for (int k = 0; k < K / 4; ++k) {
    float4 av = a4[k], wv = w4[k];
    s += av.x * wv.x + av.y * wv.y + av.z * wv.z + av.w * wv.w;
  }
  out[(long)b * sb + (long)t * st + (long)j * sn] = s + (bias ? bias[j] : 0.0f);
}

// scalar-feature embedding into concat buffer: zcat[b,base+n] = s[b]*W[n] + bias[n]
__global__ void k_emb(const float* __restrict__ s, const float* __restrict__ W,
                      const float* __restrict__ bias, float* __restrict__ zcat, int base) {
  int i = blockIdx.x * blockDim.x + threadIdx.x;
  if (i >= 64 * 512) return;
  int b = i >> 9, n = i & 511;
  zcat[b * 2048 + base + n] = s[b] * W[n] + bias[n];
}
__global__ void k_zcopy(const float* __restrict__ mu, float* __restrict__ zcat) {
  int i = blockIdx.x * blockDim.x + threadIdx.x;
  if (i >= 64 * 512) return;
  int b = i >> 9, n = i & 511;
  zcat[b * 2048 + n] = mu[i];
}

// chord decoder inputs, time-major [t][b][h]
__global__ void k_gather_chord(const float* __restrict__ z2, const int* __restrict__ gtc,
                               const float* __restrict__ cemb, float* __restrict__ cin) {
  int i = blockIdx.x * blockDim.x + threadIdx.x;
  if (i >= 32 * 64 * 512) return;
  int h = i & 511, b = (i >> 9) & 63, t = i >> 15;
  cin[i] = (t == 0) ? z2[b * 512 + h] : cemb[gtc[b * 32 + t - 1] * 512 + h];
}
// melody decoder inputs, time-major [t][b][h]
__global__ void k_gather_mel(const int* __restrict__ gtc, const int* __restrict__ gtm,
                             const float* __restrict__ cemb, const float* __restrict__ memb,
                             float* __restrict__ minb) {
  long i = (long)blockIdx.x * blockDim.x + threadIdx.x;
  if (i >= (long)512 * 64 * 512) return;
  int h = (int)(i & 511), b = (int)((i >> 9) & 63), t = (int)(i >> 15);
  float v = cemb[gtc[b * 32 + (t >> 4)] * 512 + h];
  if (t > 0) v += memb[gtm[b * 512 + t - 1] * 512 + h];
  minb[i] = v;
}

// KL = 0.5*mean(mu^2 + exp(lv) - lv - 1)
__global__ void k_kl(const float* __restrict__ mu, const float* __restrict__ lv,
                     float* __restrict__ out) {
  __shared__ float red[256];
  int tid = threadIdx.x;
  float s = 0.0f;
  for (int i = blockIdx.x * 256 + tid; i < 64 * 512; i += gridDim.x * 256) {
    float m = mu[i], l = lv[i];
    s += m * m + __expf(l) - l - 1.0f;
  }
  red[tid] = s; __syncthreads();
  for (int o = 128; o; o >>= 1) { if (tid < o) red[tid] += red[tid + o]; __syncthreads(); }
  if (tid == 0) atomicAdd(out, red[0] * (0.5f / (64.0f * 512.0f)));
}

// ---------------------------------------------------------------------------
extern "C" void kernel_launch(void* const* d_in, const int* in_sizes, int n_in,
                              void* d_out, int out_size, void* d_ws, size_t ws_size,
                              hipStream_t stream) {
  (void)in_sizes; (void)n_in; (void)out_size; (void)ws_size;
  auto F = [&](int i) { return (const float*)d_in[i]; };
  auto I = [&](int i) { return (const int*)d_in[i]; };
  float* out = (float*)d_out;
  // d_out tuple layout (flat): chord[64*32*9], melody[64*512*39], KL, bpm[64], val[64], en[64]
  const size_t O_CHORD = 0, O_MEL = 18432, O_KL = 1296384, O_BPM = 1296385, O_VAL = 1296449, O_EN = 1296513;

  // params in d_in (insertion order): enc_f{8..11} enc_b{12..15} mean{16,17} var{18,19}
  // down{20,21} bpm{22..25} val{26..29} en{30..33} bpm_emb{34,35} val_emb{36,37}
  // en_emb{38,39} chords_lstm{40..43} mel_lstm{44..47} chord_emb{48} mel_emb{49}
  // chord{50..53} mel_pred{54,55} mel_note{56,57}

  char* ws = (char*)d_ws;
  size_t off = 0;
  auto alloc = [&](size_t bytes) { size_t o = off; off = (off + bytes + 255) & ~(size_t)255; return (void*)(ws + o); };

  __bf16* wih_f = (__bf16*)alloc((size_t)2048 * 768 * 2);
  __bf16* wih_b = (__bf16*)alloc((size_t)2048 * 768 * 2);
  __bf16* whh_f = (__bf16*)alloc((size_t)2048 * 512 * 2);
  __bf16* whh_b = (__bf16*)alloc((size_t)2048 * 512 * 2);
  __bf16* wih_c = (__bf16*)alloc((size_t)2048 * 512 * 2);
  __bf16* whh_c = (__bf16*)alloc((size_t)2048 * 512 * 2);
  __bf16* wih_m = (__bf16*)alloc((size_t)2048 * 512 * 2);
  __bf16* whh_m = (__bf16*)alloc((size_t)2048 * 512 * 2);
  __bf16* w_mean = (__bf16*)alloc((size_t)512 * 512 * 2);
  __bf16* w_var  = (__bf16*)alloc((size_t)512 * 512 * 2);
  __bf16* w_down = (__bf16*)alloc((size_t)512 * 2048 * 2);
  __bf16* w_bpm1 = (__bf16*)alloc((size_t)512 * 512 * 2);
  __bf16* w_val1 = (__bf16*)alloc((size_t)512 * 512 * 2);
  __bf16* w_en1  = (__bf16*)alloc((size_t)512 * 512 * 2);
  __bf16* w_ch1  = (__bf16*)alloc((size_t)512 * 512 * 2);
  __bf16* w_mp   = (__bf16*)alloc((size_t)512 * 512 * 2);

  float* gin   = (float*)alloc((size_t)512 * 64 * 2048 * 4);  // reused: enc_f, enc_b, mel, md
  float* gin_c = (float*)alloc((size_t)32 * 64 * 2048 * 4);
  __bf16* h_ping = (__bf16*)alloc((size_t)64 * 512 * 2);
  __bf16* h_pong = (__bf16*)alloc((size_t)64 * 512 * 2);
  float* c_buf = (float*)alloc((size_t)64 * 512 * 4);
  float* hf   = (float*)alloc((size_t)64 * 512 * 4);
  float* hb   = (float*)alloc((size_t)64 * 512 * 4);
  float* temp = (float*)alloc((size_t)64 * 512 * 4);
  float* mu   = (float*)alloc((size_t)64 * 512 * 4);
  float* lv   = (float*)alloc((size_t)64 * 512 * 4);
  float* zcat = (float*)alloc((size_t)64 * 2048 * 4);
  float* z2   = (float*)alloc((size_t)64 * 512 * 4);
  float* hid  = (float*)alloc((size_t)64 * 512 * 4);
  float* cin  = (float*)alloc((size_t)32 * 64 * 512 * 4);
  float* hs_c = (float*)alloc((size_t)32 * 64 * 512 * 4);
  float* chid = (float*)alloc((size_t)32 * 64 * 512 * 4);
  float* melin = (float*)alloc((size_t)512 * 64 * 512 * 4);
  float* hs_m  = (float*)alloc((size_t)512 * 64 * 512 * 4);
  float* md = gin;  // safe reuse after melody Gin consumed

  auto cvt = [&](const float* src, __bf16* dst, int n) {
    k_f2bf<<<(n + 255) / 256, 256, 0, stream>>>(src, dst, n);
  };
  auto gemm = [&](const float* A, long sT, long sB, const __bf16* W, const float* bias,
                  float* C, int M, int N, int K, int relu) {
    dim3 g(N / 64, M / 64);
    k_gemm<<<g, 256, 0, stream>>>(A, sT, sB, W, bias, C, K, relu);
  };
  auto zeroh = [&]() {
    k_zero_bf<<<(64 * 512 + 255) / 256, 256, 0, stream>>>(h_ping, 64 * 512);
    k_zero_f<<<(64 * 512 + 255) / 256, 256, 0, stream>>>(c_buf, 64 * 512);
  };
  auto run_lstm = [&](int T, const float* ginbuf, int rev, const __bf16* whh,
                      const float* bhh, float* hs, long hs_stride) {
    for (int t = 0; t < T; ++t) {
      const __bf16* hi = (t & 1) ? h_pong : h_ping;
      __bf16* ho = (t & 1) ? h_ping : h_pong;
      int s = rev ? (T - 1 - t) : t;
      k_lstm_step<<<32, 256, 0, stream>>>(hi, ho, c_buf, whh,
          ginbuf + (long)s * 64 * 2048, bhh, hs + (long)t * hs_stride);
    }
  };

  // --- weight conversion to bf16 ---
  cvt(F(8),  wih_f, 2048 * 768);  cvt(F(12), wih_b, 2048 * 768);
  cvt(F(9),  whh_f, 2048 * 512);  cvt(F(13), whh_b, 2048 * 512);
  cvt(F(40), wih_c, 2048 * 512);  cvt(F(41), whh_c, 2048 * 512);
  cvt(F(44), wih_m, 2048 * 512);  cvt(F(45), whh_m, 2048 * 512);
  cvt(F(16), w_mean, 512 * 512);  cvt(F(18), w_var, 512 * 512);
  cvt(F(20), w_down, 512 * 2048);
  cvt(F(22), w_bpm1, 512 * 512);  cvt(F(26), w_val1, 512 * 512);  cvt(F(30), w_en1, 512 * 512);
  cvt(F(50), w_ch1, 512 * 512);   cvt(F(54), w_mp, 512 * 512);

  // --- encoder forward direction ---
  gemm(F(0), 768, (long)512 * 768, wih_f, F(10), gin, 32768, 2048, 768, 0);
  zeroh();
  run_lstm(512, gin, 0, whh_f, F(11), hf, 0);
  // --- encoder backward direction ---
  gemm(F(0), 768, (long)512 * 768, wih_b, F(14), gin, 32768, 2048, 768, 0);
  zeroh();
  run_lstm(512, gin, 1, whh_b, F(15), hb, 0);

  // temp = hf + hb ; mu / logvar
  k_add<<<(64 * 512 + 255) / 256, 256, 0, stream>>>(hf, hb, temp, 64 * 512);
  gemm(temp, 0, 512, w_mean, F(17), mu, 64, 512, 512, 0);
  gemm(temp, 0, 512, w_var,  F(19), lv, 64, 512, 512, 0);
  k_zero_f<<<1, 256, 0, stream>>>(out + O_KL, 1);
  k_kl<<<32, 256, 0, stream>>>(mu, lv, out + O_KL);

  // bpm / valence / energy heads (z = mu)
  gemm(mu, 0, 512, w_bpm1, F(23), hid, 64, 512, 512, 1);
  k_proj<<<1, 256, 0, stream>>>(hid, F(24), F(25), out + O_BPM, 64, 1, 512, 1, 0, 0);
  gemm(mu, 0, 512, w_val1, F(27), hid, 64, 512, 512, 1);
  k_proj<<<1, 256, 0, stream>>>(hid, F(28), F(29), out + O_VAL, 64, 1, 512, 1, 0, 0);
  gemm(mu, 0, 512, w_en1, F(31), hid, 64, 512, 512, 1);
  k_proj<<<1, 256, 0, stream>>>(hid, F(32), F(33), out + O_EN, 64, 1, 512, 1, 0, 0);

  // zcat = [mu, be, ve, ee] ; z2 = zcat @ down_W^T + down_b
  k_zcopy<<<(64 * 512 + 255) / 256, 256, 0, stream>>>(mu, zcat);
  k_emb<<<(64 * 512 + 255) / 256, 256, 0, stream>>>(out + O_BPM, F(34), F(35), zcat, 512);
  k_emb<<<(64 * 512 + 255) / 256, 256, 0, stream>>>(out + O_VAL, F(36), F(37), zcat, 1024);
  k_emb<<<(64 * 512 + 255) / 256, 256, 0, stream>>>(out + O_EN,  F(38), F(39), zcat, 1536);
  gemm(zcat, 0, 2048, w_down, F(21), z2, 64, 512, 2048, 0);

  // --- chord decoder ---
  k_gather_chord<<<(32 * 64 * 512 + 255) / 256, 256, 0, stream>>>(z2, I(2), F(48), cin);
  gemm(cin, (long)64 * 512, 512, wih_c, F(42), gin_c, 2048, 2048, 512, 0);
  cvt(F(4), h_ping, 64 * 512);
  k_copy_f<<<(64 * 512 + 255) / 256, 256, 0, stream>>>(F(5), c_buf, 64 * 512);
  run_lstm(32, gin_c, 0, whh_c, F(43), hs_c, (long)64 * 512);
  gemm(hs_c, (long)64 * 512, 512, w_ch1, F(51), chid, 2048, 512, 512, 1);
  k_proj<<<(2048 * 9 + 255) / 256, 256, 0, stream>>>(chid, F(52), F(53), out + O_CHORD,
                                                     2048, 9, 512, 32 * 9, 9, 1);

  // --- melody decoder ---
  {
    long n = (long)512 * 64 * 512;
    k_gather_mel<<<(unsigned)((n + 255) / 256), 256, 0, stream>>>(I(2), I(3), F(48), F(49), melin);
  }
  gemm(melin, (long)64 * 512, 512, wih_m, F(46), gin, 32768, 2048, 512, 0);
  cvt(F(6), h_ping, 64 * 512);
  k_copy_f<<<(64 * 512 + 255) / 256, 256, 0, stream>>>(F(7), c_buf, 64 * 512);
  run_lstm(512, gin, 0, whh_m, F(47), hs_m, (long)64 * 512);
  gemm(hs_m, (long)64 * 512, 512, w_mp, F(55), md, 32768, 512, 512, 1);
  k_proj<<<(32768 * 39 + 255) / 256, 256, 0, stream>>>(md, F(56), F(57), out + O_MEL,
                                                       32768, 39, 512, 512 * 39, 39, 1);
}
